// PTBlock_1726576853308
// MI455X (gfx1250) — matrix operand order repaired and checked by reading, not compiled, not run on hardware
//
#include <hip/hip_runtime.h>

// ---------------- problem constants ----------------
#define BB   8
#define NN   2048
#define CCH  128
#define CIN  64
#define KNB  16
#define C4   512
#define BN   (BB * NN)

typedef __attribute__((ext_vector_type(16))) __bf16 v16bf;
typedef __attribute__((ext_vector_type(8)))  __bf16 v8bf;
typedef __attribute__((ext_vector_type(8)))  float  v8f;
typedef unsigned short bfs;   // bf16 bit storage

__device__ __forceinline__ bfs f2bf(float f) {
  unsigned u = __builtin_bit_cast(unsigned, f);
  return (bfs)((u + 0x7FFFu + ((u >> 16) & 1u)) >> 16);   // RNE
}
__device__ __forceinline__ float bf2f(bfs h) {
  unsigned u = ((unsigned)h) << 16;
  return __builtin_bit_cast(float, u);
}
__device__ __forceinline__ __bf16 asbf(bfs h) { return __builtin_bit_cast(__bf16, h); }

__device__ __forceinline__ v8f wmma_bf16(v16bf a, v16bf b, v8f c) {
  // D = A(16x32 bf16) * B(32x16 bf16) + C(16x16 f32)
  return __builtin_amdgcn_wmma_f32_16x16x32_bf16(false, a, false, b, (short)0, c,
                                                 false, false);
}

// A fragment (16x32, lane=row): two contiguous 8-half runs at +0 and +16 halves.
__device__ __forceinline__ v16bf ld_afrag(const bfs* p0) {
  v8bf lo = *(const v8bf*)p0;         // K = kbA .. kbA+7
  v8bf hi = *(const v8bf*)(p0 + 16);  // K = kbA+16 .. kbA+23
  return __builtin_shufflevector(lo, hi, 0, 1, 2, 3, 4, 5, 6, 7,
                                 8, 9, 10, 11, 12, 13, 14, 15);
}
// B fragment (32x16, lane=col): two contiguous 8-half runs at +0 and +16 halves.
__device__ __forceinline__ v16bf ld_bfrag(const bfs* p0) {
  v8bf lo = *(const v8bf*)p0;
  v8bf hi = *(const v8bf*)(p0 + 16);
  return __builtin_shufflevector(lo, hi, 0, 1, 2, 3, 4, 5, 6, 7,
                                 8, 9, 10, 11, 12, 13, 14, 15);
}

// ---------------- weight f32 -> bf16 ----------------
__global__ void cvt_bf16_kernel(const float* __restrict__ src, bfs* __restrict__ dst, int n) {
  int i = blockIdx.x * blockDim.x + threadIdx.x;
  if (i < n) dst[i] = f2bf(src[i]);
}

// ---------------- kernel 1: h = Win @ x + bin  ([BN,128]) ----------------
__global__ __launch_bounds__(256) void inproj_kernel(
    const float* __restrict__ x, const bfs* __restrict__ Winb,
    const float* __restrict__ bin, float* __restrict__ h) {
  const int wid   = (blockIdx.x * blockDim.x + threadIdx.x) >> 5;
  const int lane  = threadIdx.x & 31;
  const int ntile = wid & 7;      // 8 col tiles of C=128
  const int mtile = wid >> 3;     // 1024 row tiles of BN
  const int row   = lane & 15;
  const int r     = mtile * 16 + row;
  const int b     = r >> 11;
  const int n     = r & (NN - 1);
  const float* xb = x + (size_t)b * CIN * NN;
  const int kbA = (lane < 16) ? 0 : 8;
  const int kbB = (lane < 16) ? 0 : 8;
  const int col = ntile * 16 + row;
  v8f acc = {};
  for (int kt = 0; kt < 2; ++kt) {
    v16bf a;
#pragma unroll
    for (int i = 0; i < 16; ++i) {
      const int kA = kt * 32 + kbA + i + ((i >= 8) ? 8 : 0);
      a[i] = asbf(f2bf(xb[(size_t)kA * NN + n]));    // strided gather of x^T
    }
    const v16bf bm = ld_bfrag(Winb + (size_t)col * CIN + kt * 32 + kbB);
    acc = wmma_bf16(a, bm, acc);
  }
  const float bias = bin[col];
  const int m0 = mtile * 16 + ((lane < 16) ? 0 : 8);
#pragma unroll
  for (int j = 0; j < 8; ++j)
    h[(size_t)(m0 + j) * CCH + col] = acc[j] + bias;
}

// ---------------- LayerNorm: one wave per point (C=128) -> bf16 ----------------
__global__ __launch_bounds__(256) void ln_kernel(
    const float* __restrict__ hin, const float* __restrict__ g,
    const float* __restrict__ bta, bfs* __restrict__ outb) {
  const int wave = threadIdx.x >> 5;
  const int lane = threadIdx.x & 31;
  const int pnt  = blockIdx.x * 8 + wave;
  const float* hp = hin + (size_t)pnt * CCH;
  float v[4];
  float s = 0.f;
#pragma unroll
  for (int i = 0; i < 4; ++i) { v[i] = hp[lane + 32 * i]; s += v[i]; }
#pragma unroll
  for (int m = 16; m >= 1; m >>= 1) s += __shfl_xor(s, m, 32);
  const float mean = s * (1.f / 128.f);
  float vs = 0.f;
#pragma unroll
  for (int i = 0; i < 4; ++i) { const float d = v[i] - mean; vs += d * d; }
#pragma unroll
  for (int m = 16; m >= 1; m >>= 1) vs += __shfl_xor(vs, m, 32);
  const float rs = rsqrtf(vs * (1.f / 128.f) + 1e-5f);
#pragma unroll
  for (int i = 0; i < 4; ++i) {
    const int c = lane + 32 * i;
    outb[(size_t)pnt * CCH + c] = f2bf((v[i] - mean) * rs * g[c] + bta[c]);
  }
}

// ---------------- KNN: 1 thread/point, LDS point tiles, reg top-16 ----------------
__global__ __launch_bounds__(256) void knn_kernel(const float* __restrict__ p,
                                                  int* __restrict__ idx) {
  const int b = blockIdx.x >> 3;
  const int n = ((blockIdx.x & 7) << 8) + threadIdx.x;
  const float* pb = p + (size_t)b * 3 * NN;
  const float qx = pb[n], qy = pb[NN + n], qz = pb[2 * NN + n];
  float bestd[KNB];
  int   besti[KNB];
#pragma unroll
  for (int t = 0; t < KNB; ++t) { bestd[t] = 1e30f; besti[t] = 0; }
  __shared__ float sx[256], sy[256], sz[256];
  for (int chunk = 0; chunk < NN; chunk += 256) {
    __syncthreads();
    const int m = chunk + threadIdx.x;
    sx[threadIdx.x] = pb[m];
    sy[threadIdx.x] = pb[NN + m];
    sz[threadIdx.x] = pb[2 * NN + m];
    __syncthreads();
    for (int t = 0; t < 256; ++t) {
      const int m2 = chunk + t;
      const float dx = qx - sx[t], dy = qy - sy[t], dz = qz - sz[t];
      const float d2 = dx * dx + dy * dy + dz * dz;
      if (m2 != n && d2 < bestd[KNB - 1]) {
        bestd[KNB - 1] = d2; besti[KNB - 1] = m2;
#pragma unroll
        for (int u = KNB - 1; u > 0; --u) {
          if (bestd[u] < bestd[u - 1]) {
            const float td = bestd[u]; bestd[u] = bestd[u - 1]; bestd[u - 1] = td;
            const int ti = besti[u]; besti[u] = besti[u - 1]; besti[u - 1] = ti;
          }
        }
      }
    }
  }
#pragma unroll
  for (int t = 0; t < KNB; ++t)
    idx[(size_t)(b * NN + n) * KNB + t] = besti[t];
}

// ---------------- QKV: one wave per 16x16 tile, shared A fragment ----------------
__global__ __launch_bounds__(256) void qkv_kernel(
    const bfs* __restrict__ hn, const bfs* __restrict__ Wqb,
    const bfs* __restrict__ Wkb, const bfs* __restrict__ Wvb,
    float* __restrict__ q, bfs* __restrict__ kall, float* __restrict__ vall) {
  const int wid   = (blockIdx.x * blockDim.x + threadIdx.x) >> 5;
  const int lane  = threadIdx.x & 31;
  const int ntile = wid & 7;
  const int mtile = wid >> 3;
  const int row   = lane & 15;
  const int kbA = (lane < 16) ? 0 : 8;
  const int kbB = (lane < 16) ? 0 : 8;
  const int col = ntile * 16 + row;
  const bfs* arow = hn + (size_t)(mtile * 16 + row) * CCH + kbA;
  const bfs* wq = Wqb + (size_t)col * CCH + kbB;
  const bfs* wk = Wkb + (size_t)col * CCH + kbB;
  const bfs* wv = Wvb + (size_t)col * CCH + kbB;
  v8f cq = {}, ck = {}, cv = {};
#pragma unroll
  for (int kt = 0; kt < 4; ++kt) {
    const v16bf a = ld_afrag(arow + kt * 32);
    cq = wmma_bf16(a, ld_bfrag(wq + kt * 32), cq);
    ck = wmma_bf16(a, ld_bfrag(wk + kt * 32), ck);
    cv = wmma_bf16(a, ld_bfrag(wv + kt * 32), cv);
  }
  const int m0 = mtile * 16 + ((lane < 16) ? 0 : 8);
#pragma unroll
  for (int j = 0; j < 8; ++j) {
    const size_t off = (size_t)(m0 + j) * CCH + col;
    q[off]    = cq[j];
    kall[off] = f2bf(ck[j]);
    vall[off] = cv[j];
  }
}

// ---------------- attention: one wave per point (16 nbrs == one M tile) ----------
__global__ __launch_bounds__(64) void attn_kernel(
    const float* __restrict__ p, const int* __restrict__ idx,
    const float* __restrict__ Wd1, const float* __restrict__ bd1,
    const bfs* __restrict__ Wd2b, const float* __restrict__ bd2,
    const bfs* __restrict__ Wab, const float* __restrict__ ba,
    const float* __restrict__ q, const bfs* __restrict__ kall,
    const float* __restrict__ vall, const float* __restrict__ h,
    float* __restrict__ hacc) {
  __shared__ int sIdx[2][KNB];
  __shared__ __align__(32) bfs sT [2][KNB * CCH];   // t, A of d-GEMM
  __shared__ __align__(32) bfs sT2[2][KNB * CCH];   // a2 = q - k_j + d
  __shared__ float sD[2][KNB * CCH];                // d (kept for y)

  const int wave = threadIdx.x >> 5;
  const int lane = threadIdx.x & 31;
  const int pnt  = blockIdx.x * 2 + wave;
  const int bb   = pnt >> 11;
  const int n    = pnt & (NN - 1);

  if (lane < KNB) sIdx[wave][lane] = idx[(size_t)pnt * KNB + lane];
  __syncthreads();

  // t = relu(rel @ Wd1^T + bd1) -> bf16 LDS (packed dword stores)
  {
    const int j   = lane & 15;
    const int nbr = sIdx[wave][j];
    const float* pb = p + (size_t)bb * 3 * NN;
    const float rx = pb[n]          - pb[nbr];
    const float ry = pb[NN + n]     - pb[NN + nbr];
    const float rz = pb[2 * NN + n] - pb[2 * NN + nbr];
    const int c0 = (lane < 16) ? 0 : 64;
    unsigned* dst = (unsigned*)&sT[wave][j * CCH];
    for (int cc = 0; cc < 64; cc += 2) {
      const int c = c0 + cc;
      const float t0 = rx * Wd1[c * 3 + 0] + ry * Wd1[c * 3 + 1] +
                       rz * Wd1[c * 3 + 2] + bd1[c];
      const float t1 = rx * Wd1[c * 3 + 3] + ry * Wd1[c * 3 + 4] +
                       rz * Wd1[c * 3 + 5] + bd1[c + 1];
      const unsigned pk = (unsigned)f2bf(fmaxf(t0, 0.0f)) |
                          ((unsigned)f2bf(fmaxf(t1, 0.0f)) << 16);
      dst[c >> 1] = pk;
    }
  }
  __syncthreads();

  const int row = lane & 15;
  const int kbA = (lane < 16) ? 0 : 8;
  const int kbB = (lane < 16) ? 0 : 8;
  const int m0  = (lane < 16) ? 0 : 8;
  const bfs* aT  = &sT[wave][row * CCH + kbA];
  const bfs* aT2 = &sT2[wave][row * CCH + kbA];

  // d = t @ Wd2^T + bd2 ; a2 = q - k_j + d
  for (int nt = 0; nt < 8; ++nt) {
    v8f acc = {};
    const int col = nt * 16 + row;
    const bfs* wd = Wd2b + (size_t)col * CCH + kbB;
#pragma unroll
    for (int kt = 0; kt < 4; ++kt)
      acc = wmma_bf16(ld_afrag(aT + kt * 32), ld_bfrag(wd + kt * 32), acc);
    const float qv  = q[(size_t)pnt * CCH + col];
    const float b2v = bd2[col];
#pragma unroll
    for (int j = 0; j < 8; ++j) {
      const int m = m0 + j;
      const float dv = acc[j] + b2v;
      sD[wave][m * CCH + col] = dv;
      const float kv = bf2f(kall[(size_t)(bb * NN + sIdx[wave][m]) * CCH + col]);
      sT2[wave][m * CCH + col] = f2bf(qv - kv + dv);
    }
  }
  __syncthreads();

  // logits = a2 @ Wa^T + ba ; softmax over neighbors (rows) ; y = sum a*(v+d)
  for (int nt = 0; nt < 8; ++nt) {
    v8f lg = {};
    const int col = nt * 16 + row;
    const bfs* wa = Wab + (size_t)col * CCH + kbB;
#pragma unroll
    for (int kt = 0; kt < 4; ++kt)
      lg = wmma_bf16(ld_afrag(aT2 + kt * 32), ld_bfrag(wa + kt * 32), lg);
    const float bav = ba[col];
    float e[8];
    float mx = -1e30f;
#pragma unroll
    for (int j = 0; j < 8; ++j) { e[j] = lg[j] + bav; mx = fmaxf(mx, e[j]); }
    mx = fmaxf(mx, __shfl_xor(mx, 16, 32));      // rows 0-7 <-> 8-15 (lane pair)
    float se = 0.0f;
#pragma unroll
    for (int j = 0; j < 8; ++j) { e[j] = __expf(e[j] - mx); se += e[j]; }
    se += __shfl_xor(se, 16, 32);
    const float inv = 1.0f / se;
    float y = 0.0f;
#pragma unroll
    for (int j = 0; j < 8; ++j) {
      const int m = m0 + j;
      const float vv = vall[(size_t)(bb * NN + sIdx[wave][m]) * CCH + col] +
                       sD[wave][m * CCH + col];
      y += e[j] * inv * vv;
    }
    y += __shfl_xor(y, 16, 32);
    if (lane < 16)
      hacc[(size_t)pnt * CCH + col] = h[(size_t)pnt * CCH + col] + y;
  }
}

// ---------------- FFN1: f = relu(h2 @ Wf1^T + bf1)  [BN,512] bf16 ----------------
__global__ __launch_bounds__(256) void ffn1_kernel(
    const bfs* __restrict__ h2, const bfs* __restrict__ W1b,
    const float* __restrict__ bf1, bfs* __restrict__ f) {
  const int wid   = (blockIdx.x * blockDim.x + threadIdx.x) >> 5;
  const int lane  = threadIdx.x & 31;
  const int ntile = wid & 31;          // 32 col tiles of 512
  const int mtile = wid >> 5;          // 1024 row tiles
  const int row   = lane & 15;
  const int kbA = (lane < 16) ? 0 : 8;
  const int kbB = (lane < 16) ? 0 : 8;
  const int col = ntile * 16 + row;
  const bfs* arow = h2 + (size_t)(mtile * 16 + row) * CCH + kbA;
  const bfs* wr   = W1b + (size_t)col * CCH + kbB;
  v8f acc = {};
#pragma unroll
  for (int kt = 0; kt < 4; ++kt)
    acc = wmma_bf16(ld_afrag(arow + kt * 32), ld_bfrag(wr + kt * 32), acc);
  const float bias = bf1[col];
  const int m0 = mtile * 16 + ((lane < 16) ? 0 : 8);
#pragma unroll
  for (int j = 0; j < 8; ++j)
    f[(size_t)(m0 + j) * C4 + col] = f2bf(fmaxf(acc[j] + bias, 0.0f));
}

// ---------------- FFN2 + residual + transpose-out  [B,C,N] ----------------
__global__ __launch_bounds__(256) void ffn2_kernel(
    const bfs* __restrict__ f, const bfs* __restrict__ W2b,
    const float* __restrict__ bf2, const float* __restrict__ hacc,
    float* __restrict__ out) {
  const int wid   = (blockIdx.x * blockDim.x + threadIdx.x) >> 5;
  const int lane  = threadIdx.x & 31;
  const int ntile = wid & 7;
  const int mtile = wid >> 3;
  const int row   = lane & 15;
  const int kbA = (lane < 16) ? 0 : 8;
  const int kbB = (lane < 16) ? 0 : 8;
  const int col = ntile * 16 + row;
  const bfs* arow = f + (size_t)(mtile * 16 + row) * C4 + kbA;
  const bfs* wr   = W2b + (size_t)col * C4 + kbB;
  v8f acc = {};
#pragma unroll
  for (int kt = 0; kt < 16; ++kt)      // K = 512
    acc = wmma_bf16(ld_afrag(arow + kt * 32), ld_bfrag(wr + kt * 32), acc);
  const float bias = bf2[col];
  const int m0 = mtile * 16 + ((lane < 16) ? 0 : 8);
#pragma unroll
  for (int j = 0; j < 8; ++j) {
    const int r = m0 + j;
    const int b = r >> 11;
    const int n = r & (NN - 1);
    out[(size_t)b * CCH * NN + (size_t)col * NN + n] =
        hacc[(size_t)r * CCH + col] + acc[j] + bias;
  }
}

// ---------------- host side ----------------
extern "C" void kernel_launch(void* const* d_in, const int* in_sizes, int n_in,
                              void* d_out, int out_size, void* d_ws, size_t ws_size,
                              hipStream_t stream) {
  const float* x   = (const float*)d_in[0];
  const float* p   = (const float*)d_in[1];
  const float* Win = (const float*)d_in[2];
  const float* bin = (const float*)d_in[3];
  const float* Wq  = (const float*)d_in[4];
  const float* Wk  = (const float*)d_in[5];
  const float* Wv  = (const float*)d_in[6];
  const float* Wd1 = (const float*)d_in[7];
  const float* bd1 = (const float*)d_in[8];
  const float* Wd2 = (const float*)d_in[9];
  const float* bd2 = (const float*)d_in[10];
  const float* Wa  = (const float*)d_in[11];
  const float* ba  = (const float*)d_in[12];
  const float* g1  = (const float*)d_in[13];
  const float* b1  = (const float*)d_in[14];
  const float* g2  = (const float*)d_in[15];
  const float* b2  = (const float*)d_in[16];
  const float* Wf1 = (const float*)d_in[17];
  const float* bf1 = (const float*)d_in[18];
  const float* Wf2 = (const float*)d_in[19];
  const float* bf2 = (const float*)d_in[20];
  float* out = (float*)d_out;

  char* ws = (char*)d_ws;
  size_t off = 0;
  auto take = [&](size_t bytes) -> char* {
    char* ptr = ws + off;
    off += (bytes + 255) & ~(size_t)255;
    return ptr;
  };
  float* h    = (float*)take((size_t)BN * CCH * 4);
  float* hacc = (float*)take((size_t)BN * CCH * 4);
  float* q    = (float*)take((size_t)BN * CCH * 4);
  float* vall = (float*)take((size_t)BN * CCH * 4);
  bfs*   hn   = (bfs*)take((size_t)BN * CCH * 2);
  bfs*   kall = (bfs*)take((size_t)BN * CCH * 2);
  bfs*   h2b  = (bfs*)take((size_t)BN * CCH * 2);
  bfs*   fbuf = (bfs*)take((size_t)BN * C4 * 2);
  int*   idxb = (int*)take((size_t)BN * KNB * 4);
  bfs* Winb = (bfs*)take((size_t)CCH * CIN * 2);
  bfs* Wqb  = (bfs*)take((size_t)CCH * CCH * 2);
  bfs* Wkb  = (bfs*)take((size_t)CCH * CCH * 2);
  bfs* Wvb  = (bfs*)take((size_t)CCH * CCH * 2);
  bfs* Wd2b = (bfs*)take((size_t)CCH * CCH * 2);
  bfs* Wab  = (bfs*)take((size_t)CCH * CCH * 2);
  bfs* Wf1b = (bfs*)take((size_t)C4 * CCH * 2);
  bfs* Wf2b = (bfs*)take((size_t)CCH * C4 * 2);

  auto cvt = [&](const float* s, bfs* d, int n) {
    cvt_bf16_kernel<<<(n + 255) / 256, 256, 0, stream>>>(s, d, n);
  };
  cvt(Win, Winb, CCH * CIN);
  cvt(Wq,  Wqb,  CCH * CCH);
  cvt(Wk,  Wkb,  CCH * CCH);
  cvt(Wv,  Wvb,  CCH * CCH);
  cvt(Wd2, Wd2b, CCH * CCH);
  cvt(Wa,  Wab,  CCH * CCH);
  cvt(Wf1, Wf1b, C4 * CCH);
  cvt(Wf2, Wf2b, CCH * C4);

  inproj_kernel<<<1024, 256, 0, stream>>>(x, Winb, bin, h);
  ln_kernel<<<BN / 8, 256, 0, stream>>>(h, g1, b1, hn);
  knn_kernel<<<BB * (NN / 256), 256, 0, stream>>>(p, idxb);
  qkv_kernel<<<1024, 256, 0, stream>>>(hn, Wqb, Wkb, Wvb, q, kall, vall);
  attn_kernel<<<BN / 2, 64, 0, stream>>>(p, idxb, Wd1, bd1, Wd2b, bd2, Wab, ba,
                                         q, kall, vall, h, hacc);
  ln_kernel<<<BN / 8, 256, 0, stream>>>(hacc, g2, b2, h2b);
  ffn1_kernel<<<4096, 256, 0, stream>>>(h2b, Wf1b, bf1, fbuf);
  ffn2_kernel<<<1024, 256, 0, stream>>>(fbuf, Wf2b, bf2, hacc, out);

  (void)in_sizes; (void)n_in; (void)out_size; (void)ws_size;
}
